// Causal_46084999086801
// MI455X (gfx1250) — matrix operand order repaired and checked by reading, not compiled, not run on hardware
//
#include <hip/hip_runtime.h>
#include <hip/hip_bf16.h>

typedef __attribute__((ext_vector_type(16))) _Float16 v16h;
typedef __attribute__((ext_vector_type(8)))  _Float16 v8h;
typedef __attribute__((ext_vector_type(8)))  float    v8f;

#define HDIM 128
#define GNUM 128
#define CNUM 10
#define BN_EPS 1e-5f

// ---------------- elementwise utilities ----------------
__global__ void k_zero(float* p, long n) {
  long i = (long)blockIdx.x * blockDim.x + threadIdx.x;
  if (i < n) p[i] = 0.f;
}
__global__ void k_fill(float* p, float v, long n) {
  long i = (long)blockIdx.x * blockDim.x + threadIdx.x;
  if (i < n) p[i] = v;
}
__global__ void k_copy(float* __restrict__ d, const float* __restrict__ s, long n) {
  long i = (long)blockIdx.x * blockDim.x + threadIdx.x;
  if (i < n) d[i] = s[i];
}

// ---------------- BatchNorm (training mode, biased var), 128 columns ----------------
__global__ void k_bn_stats(const float* __restrict__ X, int R,
                           float* __restrict__ sum, float* __restrict__ sumsq) {
  __shared__ float s1[256], s2[256];
  const int c    = threadIdx.x & 127;
  const int half = threadIdx.x >> 7;
  const long r0  = (long)blockIdx.x * 128;
  long rend = r0 + 128; if (rend > R) rend = R;
  float a = 0.f, b = 0.f;
  for (long r = r0 + half; r < rend; r += 2) {
    float v = X[r * HDIM + c];
    a += v; b += v * v;
  }
  s1[threadIdx.x] = a; s2[threadIdx.x] = b;
  __syncthreads();
  if (threadIdx.x < 128) {
    atomicAdd(&sum[c],   s1[threadIdx.x] + s1[threadIdx.x + 128]);
    atomicAdd(&sumsq[c], s2[threadIdx.x] + s2[threadIdx.x + 128]);
  }
}
__global__ void k_bn_finalize(const float* __restrict__ sum, const float* __restrict__ sumsq,
                              float R, float* __restrict__ mean, float* __restrict__ rstd) {
  int c = threadIdx.x;
  if (c < 128) {
    float m = sum[c] / R;
    float v = sumsq[c] / R - m * m;
    if (v < 0.f) v = 0.f;
    mean[c] = m;
    rstd[c] = rsqrtf(v + BN_EPS);
  }
}
__global__ void k_bn_apply(const float* __restrict__ X, const float* __restrict__ mean,
                           const float* __restrict__ rstd, const float* __restrict__ g,
                           const float* __restrict__ b, float* __restrict__ Y,
                           long n, int relu) {
  long i = (long)blockIdx.x * blockDim.x + threadIdx.x;
  if (i >= n) return;
  int c = (int)(i & 127);
  float v = (X[i] - mean[c]) * rstd[c] * g[c] + b[c];
  if (relu) v = fmaxf(v, 0.f);
  Y[i] = v;
}

// ---------------- WMMA GEMM: Out(Mx128) = A(Mx128) @ W(128x128) + bias, opt ReLU ----
// block = 256 threads (8 wave32), covers 16 rows x 128 cols; grid.x = M/16.
__global__ __launch_bounds__(256) void k_gemm_wmma(const float* __restrict__ A,
                                                   const float* __restrict__ W,
                                                   const float* __restrict__ bias,
                                                   float* __restrict__ Out, int relu) {
  __shared__ _Float16 lA[16 * HDIM];        // A tile, row-major f16
  __shared__ _Float16 lBt[HDIM * HDIM];     // W transposed: lBt[n][k]
  const int tid = threadIdx.x;
  const long row0 = (long)blockIdx.x * 16;
#pragma unroll
  for (int i = 0; i < 8; ++i) {             // 2048 elems of A tile
    int idx = i * 256 + tid;
    lA[idx] = (_Float16)A[(row0 + (idx >> 7)) * HDIM + (idx & 127)];
  }
#pragma unroll
  for (int i = 0; i < 64; ++i) {            // 16384 elems of W, transposed store
    int idx = i * 256 + tid;
    int k = idx >> 7, n = idx & 127;
    lBt[n * HDIM + k] = (_Float16)W[idx];
  }
  __syncthreads();

  const int lane = tid & 31;
  const int wave = tid >> 5;
  const int ml   = lane & 15;               // row-in-tile for A, col-in-tile for B
  const int hi   = lane >> 4;
  const int ncol = wave * 16 + ml;          // global output column

  v8f c = {};
#pragma unroll
  for (int ks = 0; ks < 4; ++ks) {
    // A 16x32 f16 frag: halves 0-7 -> K=kb..kb+7, halves 8-15 -> K=kb+16..kb+23, kb=ks*32+hi*8
    const int kbA = ks * 32 + hi * 8;
    const v8h a0 = *(const v8h*)&lA[ml * HDIM + kbA];
    const v8h a1 = *(const v8h*)&lA[ml * HDIM + kbA + 16];
    // B 32x16 f16 frag: halves 0-7 -> K=kb..kb+7, halves 8-15 -> K=kb+16..kb+23, kb=ks*32+hi*8
    const int kbB = ks * 32 + hi * 8;
    const v8h b0 = *(const v8h*)&lBt[ncol * HDIM + kbB];
    const v8h b1 = *(const v8h*)&lBt[ncol * HDIM + kbB + 16];
    v16h av, bv;
#pragma unroll
    for (int j = 0; j < 8; ++j) {
      av[j] = a0[j]; av[j + 8] = a1[j];
      bv[j] = b0[j]; bv[j + 8] = b1[j];
    }
    c = __builtin_amdgcn_wmma_f32_16x16x32_f16(false, av, false, bv, (short)0, c,
                                               false, false);
  }
  const float bb = bias ? bias[ncol] : 0.f;
#pragma unroll
  for (int r = 0; r < 8; ++r) {             // C/D: VGPR r -> M=r (lanes 0-15), r+8 (16-31)
    long mrow = row0 + r + hi * 8;
    float v = c[r] + bb;
    if (relu) v = fmaxf(v, 0.f);
    Out[mrow * HDIM + ncol] = v;
  }
}

// ---------------- edge kernels ----------------
__global__ void k_edge_scatter(const float* __restrict__ h, const int* __restrict__ row,
                               const int* __restrict__ col, float* __restrict__ agg, int E) {
  int e = blockIdx.x * 8 + (threadIdx.x >> 5);
  if (e >= E) return;
  int lane = threadIdx.x & 31;
  long r = row[e], c = col[e];
  const float4 v = *(const float4*)&h[r * HDIM + lane * 4];
  float* dst = &agg[c * HDIM + lane * 4];
  atomicAdd(dst + 0, v.x); atomicAdd(dst + 1, v.y);
  atomicAdd(dst + 2, v.z); atomicAdd(dst + 3, v.w);
}

__global__ void k_node_proj(const float* __restrict__ h, const float* __restrict__ We,
                            const float* __restrict__ Wna, const float* __restrict__ bna,
                            float* __restrict__ np_, float* __restrict__ na_, int N) {
  int n = blockIdx.x * 8 + (threadIdx.x >> 5);
  if (n >= N) return;
  int lane = threadIdx.x & 31;
  const float4 hv = *(const float4*)&h[(long)n * HDIM + lane * 4];
  float hv4[4] = {hv.x, hv.y, hv.z, hv.w};
  float acc[6] = {0.f, 0.f, 0.f, 0.f, 0.f, 0.f};
#pragma unroll
  for (int j = 0; j < 4; ++j) {
    int f = lane * 4 + j;
    float x = hv4[j];
    acc[0] += x * We[f * 2 + 0];
    acc[1] += x * We[f * 2 + 1];
    acc[2] += x * We[(HDIM + f) * 2 + 0];
    acc[3] += x * We[(HDIM + f) * 2 + 1];
    acc[4] += x * Wna[f * 2 + 0];
    acc[5] += x * Wna[f * 2 + 1];
  }
#pragma unroll
  for (int s = 16; s > 0; s >>= 1)
#pragma unroll
    for (int q = 0; q < 6; ++q) acc[q] += __shfl_xor(acc[q], s, 32);
  if (lane == 0) {
    np_[(long)n * 4 + 0] = acc[0]; np_[(long)n * 4 + 1] = acc[1];
    np_[(long)n * 4 + 2] = acc[2]; np_[(long)n * 4 + 3] = acc[3];
    float l0 = acc[4] + bna[0], l1 = acc[5] + bna[1];
    float mx = fmaxf(l0, l1);
    float e0 = expf(l0 - mx), e1 = expf(l1 - mx), s = e0 + e1;
    na_[(long)n * 2 + 0] = e0 / s; na_[(long)n * 2 + 1] = e1 / s;
  }
}

__global__ void k_edge_att(const float* __restrict__ np_, const int* __restrict__ row,
                           const int* __restrict__ col, const float* __restrict__ be,
                           float* __restrict__ ea, int E) {
  int e = blockIdx.x * blockDim.x + threadIdx.x;
  if (e >= E) return;
  long r = row[e], c = col[e];
  float l0 = np_[r * 4 + 0] + np_[c * 4 + 2] + be[0];
  float l1 = np_[r * 4 + 1] + np_[c * 4 + 3] + be[1];
  float mx = fmaxf(l0, l1);
  float e0 = expf(l0 - mx), e1 = expf(l1 - mx), s = e0 + e1;
  ea[(long)e * 2 + 0] = e0 / s;
  ea[(long)e * 2 + 1] = e1 / s;
}

__global__ void k_xscale(const float* __restrict__ h, const float* __restrict__ na_,
                         int which, float* __restrict__ dst, long n) {
  long i = (long)blockIdx.x * blockDim.x + threadIdx.x;
  if (i >= n) return;
  long node = i >> 7;
  dst[i] = na_[node * 2 + which] * h[i];
}

__global__ void k_deg_scatter(const float* __restrict__ ea, int which,
                              const int* __restrict__ row, float* __restrict__ deg, int E) {
  int e = blockIdx.x * blockDim.x + threadIdx.x;
  if (e >= E) return;
  atomicAdd(&deg[row[e]], ea[(long)e * 2 + which]);
}
__global__ void k_dinv(const float* __restrict__ deg, float* __restrict__ dnv, int N) {
  int i = blockIdx.x * blockDim.x + threadIdx.x;
  if (i >= N) return;
  float d = deg[i];
  dnv[i] = (d > 0.f) ? rsqrtf(d) : 0.f;
}
__global__ void k_selfloop(const float* __restrict__ hW, const float* __restrict__ dnv,
                           float* __restrict__ out, long n) {
  long i = (long)blockIdx.x * blockDim.x + threadIdx.x;
  if (i >= n) return;
  long node = i >> 7;
  float d = dnv[node];
  out[i] = d * d * hW[i];
}
__global__ void k_gcn_scatter(const float* __restrict__ hW, const float* __restrict__ ea,
                              int which, const float* __restrict__ dnv,
                              const int* __restrict__ row, const int* __restrict__ col,
                              float* __restrict__ out, int E) {
  int e = blockIdx.x * 8 + (threadIdx.x >> 5);
  if (e >= E) return;
  int lane = threadIdx.x & 31;
  long r = row[e], c = col[e];
  float norm = dnv[r] * ea[(long)e * 2 + which] * dnv[c];
  const float4 v = *(const float4*)&hW[r * HDIM + lane * 4];
  float* dst = &out[c * HDIM + lane * 4];
  atomicAdd(dst + 0, norm * v.x); atomicAdd(dst + 1, norm * v.y);
  atomicAdd(dst + 2, norm * v.z); atomicAdd(dst + 3, norm * v.w);
}
__global__ void k_finalize_pool(const float* __restrict__ acc, const float* __restrict__ bias,
                                const int* __restrict__ batch, float* __restrict__ pool,
                                long n) {
  long i = (long)blockIdx.x * blockDim.x + threadIdx.x;
  if (i >= n) return;
  long node = i >> 7;
  int f = (int)(i & 127);
  float v = fmaxf(acc[i] + bias[f], 0.f);
  atomicAdd(&pool[(long)batch[node] * HDIM + f], v);
}

// ---------------- readout tail ----------------
__global__ void k_small_gemm(const float* __restrict__ X, const float* __restrict__ W2,
                             const float* __restrict__ b2, float* __restrict__ out, long n) {
  long idx = (long)blockIdx.x * blockDim.x + threadIdx.x;
  if (idx >= n) return;
  long r = idx / CNUM;
  int c = (int)(idx % CNUM);
  float s = 0.f;
#pragma unroll 8
  for (int k = 0; k < HDIM; ++k) s += X[r * HDIM + k] * W2[k * CNUM + c];
  out[idx] = s + b2[c];
}
__global__ void k_logsoftmax10(const float* __restrict__ in, float* __restrict__ out, int R) {
  int r = blockIdx.x * blockDim.x + threadIdx.x;
  if (r >= R) return;
  float l[CNUM];
  float mx = -1e30f;
#pragma unroll
  for (int c = 0; c < CNUM; ++c) { l[c] = in[(long)r * CNUM + c]; mx = fmaxf(mx, l[c]); }
  float s = 0.f;
#pragma unroll
  for (int c = 0; c < CNUM; ++c) s += expf(l[c] - mx);
  float lse = mx + logf(s);
#pragma unroll
  for (int c = 0; c < CNUM; ++c) out[(long)r * CNUM + c] = l[c] - lse;
}
__global__ void k_xadd(const float* __restrict__ poolO, const float* __restrict__ poolC,
                       float* __restrict__ out, long n) {
  long i = (long)blockIdx.x * blockDim.x + threadIdx.x;
  if (i >= n) return;
  int f = (int)(i & 127);
  long pr = i >> 7;            // pr = a*G + b
  long a = pr >> 7, b = pr & 127;
  out[i] = poolO[a * HDIM + f] + poolC[b * HDIM + f];
}

// ---------------- host orchestration ----------------
extern "C" void kernel_launch(void* const* d_in, const int* in_sizes, int n_in,
                              void* d_out, int out_size, void* d_ws, size_t ws_size,
                              hipStream_t stream) {
  const float* x       = (const float*)d_in[0];
  const float* bnf_g   = (const float*)d_in[1];
  const float* bnf_b   = (const float*)d_in[2];
  const float* Wfeat   = (const float*)d_in[3];
  const float* bfeat   = (const float*)d_in[4];
  const float* gin_W1  = (const float*)d_in[5];
  const float* gin_b1  = (const float*)d_in[6];
  const float* gin_bng = (const float*)d_in[7];
  const float* gin_bnb = (const float*)d_in[8];
  const float* gin_W2  = (const float*)d_in[9];
  const float* gin_b2  = (const float*)d_in[10];
  const float* We      = (const float*)d_in[11];
  const float* be      = (const float*)d_in[12];
  const float* Wna     = (const float*)d_in[13];
  const float* bna     = (const float*)d_in[14];
  const float* bnc_g   = (const float*)d_in[15];
  const float* bnc_b   = (const float*)d_in[16];
  const float* bno_g   = (const float*)d_in[17];
  const float* bno_b   = (const float*)d_in[18];
  const float* Wctx    = (const float*)d_in[19];
  const float* bctx    = (const float*)d_in[20];
  const float* Wobj    = (const float*)d_in[21];
  const float* bobj    = (const float*)d_in[22];
  const int*   row     = (const int*)d_in[47];
  const int*   col     = (const int*)d_in[48];
  const int*   batch   = (const int*)d_in[49];

  const int  N  = in_sizes[0] / HDIM;
  const int  E  = in_sizes[47];
  const long NH = (long)N * HDIM;

  float* ws    = (float*)d_ws;
  float* bufH  = ws;                 // NH   : node features h
  float* bufB  = ws + NH;            // NH   : agg / masked features
  float* bufT1 = ws + 2 * NH;        // NH   : temp
  float* bufT2 = ws + 3 * NH;        // NH   : temp
  float* m     = ws + 4 * NH;
  float* np_   = m;  m += (long)N * 4;
  float* na_   = m;  m += (long)N * 2;
  float* ea    = m;  m += (long)E * 2;
  float* deg   = m;  m += N;
  float* dnv   = m;  m += N;
  float* ssum  = m;  m += 128;       // ssum/ssq contiguous (256 floats)
  float* ssq   = m;  m += 128;
  float* meanb = m;  m += 128;
  float* rstdb = m;  m += 128;
  float* poolC = m;  m += (long)GNUM * HDIM;
  float* poolO = m;  m += (long)GNUM * HDIM;
  float* logit = m;  m += (long)GNUM * GNUM * CNUM;

  auto blk = [](long n) { return (unsigned)((n + 255) / 256); };

  auto bn_full = [&](const float* X, int R, const float* g, const float* b,
                     float* Y, int relu) {
    k_zero<<<1, 256, 0, stream>>>(ssum, 256L);
    k_bn_stats<<<(R + 127) / 128, 256, 0, stream>>>(X, R, ssum, ssq);
    k_bn_finalize<<<1, 128, 0, stream>>>(ssum, ssq, (float)R, meanb, rstdb);
    long n = (long)R * HDIM;
    k_bn_apply<<<blk(n), 256, 0, stream>>>(X, meanb, rstdb, g, b, Y, n, relu);
  };
  auto gemm = [&](const float* A, const float* W, const float* bias, float* Out,
                  int M, int relu) {
    k_gemm_wmma<<<M / 16, 256, 0, stream>>>(A, W, bias, Out, relu);
  };
  auto gcn_branch = [&](const float* src, int which, const float* bng, const float* bnb,
                        const float* W, const float* bias, float* pool) {
    bn_full(src, N, bng, bnb, bufT1, 0);
    gemm(bufT1, W, nullptr, bufT2, N, 0);                       // h @ W
    k_fill<<<blk(N), 256, 0, stream>>>(deg, 1.0f, (long)N);     // self-loop weight 1
    k_deg_scatter<<<blk(E), 256, 0, stream>>>(ea, which, row, deg, E);
    k_dinv<<<blk(N), 256, 0, stream>>>(deg, dnv, N);
    k_selfloop<<<blk(NH), 256, 0, stream>>>(bufT2, dnv, bufT1, NH);
    k_gcn_scatter<<<(E + 7) / 8, 256, 0, stream>>>(bufT2, ea, which, dnv, row, col,
                                                   bufT1, E);
    k_zero<<<blk((long)GNUM * HDIM), 256, 0, stream>>>(pool, (long)GNUM * HDIM);
    k_finalize_pool<<<blk(NH), 256, 0, stream>>>(bufT1, bias, batch, pool, NH);
  };
  auto readout = [&](const float* X, int R, int base, float* out) {
    const float* bn1g = (const float*)d_in[base + 0];
    const float* bn1b = (const float*)d_in[base + 1];
    const float* W1   = (const float*)d_in[base + 2];
    const float* b1   = (const float*)d_in[base + 3];
    const float* bn2g = (const float*)d_in[base + 4];
    const float* bn2b = (const float*)d_in[base + 5];
    const float* W2   = (const float*)d_in[base + 6];
    const float* b2   = (const float*)d_in[base + 7];
    bn_full(X, R, bn1g, bn1b, bufT1, 0);
    gemm(bufT1, W1, b1, bufT2, R, 1);
    bn_full(bufT2, R, bn2g, bn2b, bufT1, 0);
    long nl = (long)R * CNUM;
    k_small_gemm<<<blk(nl), 256, 0, stream>>>(bufT1, W2, b2, logit, nl);
    k_logsoftmax10<<<blk(R), 256, 0, stream>>>(logit, out, R);
  };

  // ---- feature BN + conv_feat (plain linear + relu) ----
  bn_full(x, N, bnf_g, bnf_b, bufT1, 0);
  gemm(bufT1, Wfeat, bfeat, bufH, N, 1);

  // ---- GIN layers ----
  for (int i = 0; i < 3; ++i) {
    k_copy<<<blk(NH), 256, 0, stream>>>(bufB, bufH, NH);                  // agg = h
    k_edge_scatter<<<(E + 7) / 8, 256, 0, stream>>>(bufH, row, col, bufB, E);
    gemm(bufB, gin_W1 + (long)i * HDIM * HDIM, gin_b1 + i * HDIM, bufT1, N, 0);
    bn_full(bufT1, N, gin_bng + i * HDIM, gin_bnb + i * HDIM, bufT2, 1);  // relu(bn)
    gemm(bufT2, gin_W2 + (long)i * HDIM * HDIM, gin_b2 + i * HDIM, bufH, N, 1);
  }

  // ---- edge + node attention ----
  k_node_proj<<<(N + 7) / 8, 256, 0, stream>>>(bufH, We, Wna, bna, np_, na_, N);
  k_edge_att<<<blk(E), 256, 0, stream>>>(np_, row, col, be, ea, E);

  // ---- two GCN branches (sequential, reuse buffers) ----
  k_xscale<<<blk(NH), 256, 0, stream>>>(bufH, na_, 0, bufB, NH);          // xc
  gcn_branch(bufB, 0, bnc_g, bnc_b, Wctx, bctx, poolC);
  k_xscale<<<blk(NH), 256, 0, stream>>>(bufH, na_, 1, bufB, NH);          // xo
  gcn_branch(bufB, 1, bno_g, bno_b, Wobj, bobj, poolO);

  // ---- readouts ----
  float* out = (float*)d_out;
  readout(poolC, GNUM, 23, out);                                  // c_
  readout(poolO, GNUM, 31, out + GNUM * CNUM);                    // o_
  long nadd = (long)GNUM * GNUM * HDIM;
  k_xadd<<<blk(nadd), 256, 0, stream>>>(poolO, poolC, bufH, nadd);
  readout(bufH, GNUM * GNUM, 39, out + 2 * GNUM * CNUM);          // co_
}
